// LinearTransformer_26079041421691
// MI455X (gfx1250) — matrix operand order repaired and checked by reading, not compiled, not run on hardware
//
#include <hip/hip_runtime.h>
#include <math.h>

// ---------------------------------------------------------------------------
// Types for CDNA5 WMMA (gfx1250, wave32)
// ---------------------------------------------------------------------------
typedef __attribute__((ext_vector_type(16))) __bf16  v16bf;
typedef __attribute__((ext_vector_type(8)))  float   v8f;
typedef __attribute__((ext_vector_type(4)))  unsigned int v4u;

union FragU { v16bf v; v4u u[2]; };   // 32B: 16 bf16 = two 16B loads

static_assert(sizeof(__bf16) == 2, "bf16 size");

#define D_MODEL 1024
#define D_FF    4096
#define NH      16
#define HD      64
#define SEQ     4096
#define BATCH   2
#define M_TOT   (BATCH * SEQ)     // 8192 rows
#define CHUNK   128
#define NCHUNK  (SEQ / CHUNK)     // 32

__device__ __forceinline__ float  bf2f(__bf16 x) { return (float)x; }
__device__ __forceinline__ __bf16 f2bf(float  x) { return (__bf16)x; }
__device__ __forceinline__ v8f zero8() { v8f z = {0.f,0.f,0.f,0.f,0.f,0.f,0.f,0.f}; return z; }

// ---------------------------------------------------------------------------
// CDNA5 async copy global->LDS (ASYNCcnt-tracked; per-lane 16B transfers)
// Generic pointer to a __shared__ object carries the LDS byte offset in its
// low 32 bits on AMDGPU, which is exactly what the VDST operand wants.
// ---------------------------------------------------------------------------
__device__ __forceinline__ unsigned lds_u32(const void* p) {
  return (unsigned)(uintptr_t)p;
}
__device__ __forceinline__ void async_b128(unsigned lds_off, const void* gaddr) {
  asm volatile("global_load_async_to_lds_b128 %0, %1, off"
               :: "v"(lds_off), "v"((unsigned long long)(uintptr_t)gaddr)
               : "memory");
}
__device__ __forceinline__ void wait_async_le8() {
  asm volatile("s_wait_asynccnt 0x8" ::: "memory");
}
__device__ __forceinline__ void wait_async_le0() {
  asm volatile("s_wait_asynccnt 0x0" ::: "memory");
}

// ---------------------------------------------------------------------------
// Utility kernels
// ---------------------------------------------------------------------------
__global__ void lt_copy_x(const float* __restrict__ xin, float* __restrict__ xf,
                          __bf16* __restrict__ xb, size_t n) {
  size_t i = (size_t)blockIdx.x * blockDim.x + threadIdx.x;
  if (i < n) { float v = xin[i]; xf[i] = v; xb[i] = f2bf(v); }
}

// Tiled transpose-convert: W [K][N] fp32 -> WT [N][K] bf16 (coalesced both ways)
// grid (N/32, K/32), block 256 (32x8)
__global__ __launch_bounds__(256)
void lt_wt_convert(const float* __restrict__ W, __bf16* __restrict__ WT,
                   int K, int N) {
  __shared__ float t[32][33];
  int n0 = blockIdx.x * 32, k0 = blockIdx.y * 32;
  int tx = threadIdx.x & 31, ty = threadIdx.x >> 5;
#pragma unroll
  for (int i = 0; i < 32; i += 8)
    t[ty + i][tx] = W[(size_t)(k0 + ty + i) * N + n0 + tx];
  __syncthreads();
#pragma unroll
  for (int i = 0; i < 32; i += 8)
    WT[(size_t)(n0 + ty + i) * K + k0 + tx] = f2bf(t[tx][ty + i]);
}

// Row LayerNorm over D_MODEL; optional bf16 mirror for next GEMM input.
__global__ __launch_bounds__(256)
void lt_layernorm(const float* __restrict__ xin, const float* __restrict__ g,
                  const float* __restrict__ be, float* __restrict__ outF,
                  __bf16* __restrict__ outB) {
  int row = blockIdx.x;
  const float* xr = xin + (size_t)row * D_MODEL;
  __shared__ float red0[8];
  __shared__ float red1[8];
  float s = 0.f, s2 = 0.f;
  for (int c = threadIdx.x; c < D_MODEL; c += 256) {
    float v = xr[c]; s += v; s2 += v * v;
  }
  for (int m = 16; m >= 1; m >>= 1) {
    s  += __shfl_xor(s,  m, 32);
    s2 += __shfl_xor(s2, m, 32);
  }
  int wave = threadIdx.x >> 5, lane = threadIdx.x & 31;
  if (lane == 0) { red0[wave] = s; red1[wave] = s2; }
  __syncthreads();
  if (threadIdx.x == 0) {
    float a = 0.f, b = 0.f;
    for (int i = 0; i < 8; i++) { a += red0[i]; b += red1[i]; }
    red0[0] = a; red1[0] = b;
  }
  __syncthreads();
  float mean = red0[0] * (1.f / D_MODEL);
  float var  = red1[0] * (1.f / D_MODEL) - mean * mean;
  float rs   = rsqrtf(var + 1e-5f);
  for (int c = threadIdx.x; c < D_MODEL; c += 256) {
    float v = (xr[c] - mean) * rs * g[c] + be[c];
    outF[(size_t)row * D_MODEL + c] = v;
    if (outB) outB[(size_t)row * D_MODEL + c] = f2bf(v);
  }
}

// ---------------------------------------------------------------------------
// WMMA bf16 GEMM with double-buffered async global->LDS pipeline.
//   C[M,N] = A[M,K] * BT[N,K]^T + bias, epilogue:
//   EPI 0: fp32 store          EPI 1: elu(x)+1 -> bf16   EPI 2: bf16 store
//   EPI 3: exact GELU -> bf16  EPI 4: resid + val -> fp32
// Block = 256 thr (8 waves), tile 128x128, K-step 64, wave = 32x64.
// ---------------------------------------------------------------------------
#define KB    64
#define LPAD  72   // 64 + 8 bf16 pad per row (16B aligned, de-conflicts banks)

template <int EPI>
__global__ __launch_bounds__(256)
void lt_gemm_bf16(const __bf16* __restrict__ A, const __bf16* __restrict__ BT,
                  const float* __restrict__ bias, const float* __restrict__ resid,
                  float* __restrict__ outF, __bf16* __restrict__ outB,
                  int M, int N, int K) {
  __shared__ __align__(16) __bf16 sA[2][128][LPAD];
  __shared__ __align__(16) __bf16 sB[2][128][LPAD];

  const int tid  = threadIdx.x;
  const int lane = tid & 31;
  const int wave = tid >> 5;
  const int lr   = lane & 15;
  const int hf   = lane >> 4;
  const int mblk = blockIdx.y * 128;
  const int nblk = blockIdx.x * 128;
  const int mbase = (wave >> 1) * 32;   // tile-local
  const int nbase = (wave & 1) * 64;

  // async copy mapping: 256 lanes x 16B = 4KB per issue; 4 issues per 16KB tile
  const int crow = tid >> 3;            // 0..31
  const int ccol = (tid & 7) * 8;       // 0,8,...,56 (bf16 elements)

  const int nIter = K / KB;

  auto issue = [&](int buf, int k0) {
#pragma unroll
    for (int i = 0; i < 4; i++) {
      int r = crow + i * 32;
      async_b128(lds_u32(&sA[buf][r][ccol]),
                 A + (size_t)(mblk + r) * K + k0 + ccol);
    }
#pragma unroll
    for (int i = 0; i < 4; i++) {
      int r = crow + i * 32;
      async_b128(lds_u32(&sB[buf][r][ccol]),
                 BT + (size_t)(nblk + r) * K + k0 + ccol);
    }
  };

  issue(0, 0);
  if (nIter > 1) issue(1, KB);

  v8f acc[2][4];
#pragma unroll
  for (int mt = 0; mt < 2; mt++)
#pragma unroll
    for (int nt = 0; nt < 4; nt++) acc[mt][nt] = zero8();

  for (int it = 0; it < nIter; it++) {
    const int buf = it & 1;
    // oldest 8 outstanding ops belong to `buf`; keep next buffer in flight
    if (it + 1 < nIter) wait_async_le8(); else wait_async_le0();
    __syncthreads();

#pragma unroll
    for (int ks = 0; ks < KB; ks += 32) {
      FragU a[2];
#pragma unroll
      for (int mt = 0; mt < 2; mt++) {
        const __bf16* p = &sA[buf][mbase + mt * 16 + lr][ks + hf * 8];
        a[mt].u[0] = *(const v4u*)p;
        a[mt].u[1] = *(const v4u*)(p + 16);
      }
      FragU b[4];
#pragma unroll
      for (int nt = 0; nt < 4; nt++) {
        const __bf16* p = &sB[buf][nbase + nt * 16 + lr][ks + hf * 8];
        b[nt].u[0] = *(const v4u*)p;
        b[nt].u[1] = *(const v4u*)(p + 16);
      }
#pragma unroll
      for (int mt = 0; mt < 2; mt++)
#pragma unroll
        for (int nt = 0; nt < 4; nt++)
          acc[mt][nt] = __builtin_amdgcn_wmma_f32_16x16x32_bf16(
              false, a[mt].v, false, b[nt].v, (short)0, acc[mt][nt], false, false);
    }

    __syncthreads();
    if (it + 2 < nIter) issue(buf, (it + 2) * KB);
  }

#pragma unroll
  for (int mt = 0; mt < 2; mt++)
#pragma unroll
    for (int nt = 0; nt < 4; nt++)
#pragma unroll
      for (int i = 0; i < 8; i++) {
        int row = mblk + mbase + mt * 16 + i + hf * 8;
        int col = nblk + nbase + nt * 16 + lr;
        size_t idx = (size_t)row * N + col;
        float v = acc[mt][nt][i] + bias[col];
        if (EPI == 0) {
          outF[idx] = v;
        } else if (EPI == 1) {              // elu(x)+1
          v = v > 0.f ? v + 1.f : __expf(v);
          outB[idx] = f2bf(v);
        } else if (EPI == 2) {
          outB[idx] = f2bf(v);
        } else if (EPI == 3) {              // exact GELU
          v = 0.5f * v * (1.f + erff(v * 0.70710678118654752f));
          outB[idx] = f2bf(v);
        } else {                             // residual add
          outF[idx] = resid[idx] + v;
        }
      }
}

// ---------------------------------------------------------------------------
// Attention phase A: per-chunk KV[d][e] = sum_c k[c][d] v[c][e]; Ks[d] = sum_c k
// grid (nC, H, B), 256 thr
// ---------------------------------------------------------------------------
__global__ __launch_bounds__(256)
void lt_kv_state(const __bf16* __restrict__ kf, const __bf16* __restrict__ vf,
                 float* __restrict__ KV, float* __restrict__ Ks) {
  int n = blockIdx.x, h = blockIdx.y, b = blockIdx.z;
  int s0 = b * SEQ + n * CHUNK;
  __shared__ float kl[CHUNK][HD];
  __shared__ float vl[CHUNK][HD];
  for (int e = threadIdx.x; e < CHUNK * HD; e += 256) {
    int r = e >> 6, c = e & 63;
    size_t gi = (size_t)(s0 + r) * D_MODEL + h * HD + c;
    kl[r][c] = bf2f(kf[gi]);
    vl[r][c] = bf2f(vf[gi]);
  }
  __syncthreads();
  int d  = threadIdx.x & 63;
  int eb = (threadIdx.x >> 6) * 16;
  float acc[16];
#pragma unroll
  for (int j = 0; j < 16; j++) acc[j] = 0.f;
  for (int c = 0; c < CHUNK; c++) {
    float kc = kl[c][d];
#pragma unroll
    for (int j = 0; j < 16; j++) acc[j] += kc * vl[c][eb + j];
  }
  size_t base = ((size_t)((b * NH + h) * NCHUNK + n)) * HD * HD;
  for (int j = 0; j < 16; j++) KV[base + (size_t)d * HD + eb + j] = acc[j];
  if (threadIdx.x < HD) {
    float s = 0.f;
    for (int c = 0; c < CHUNK; c++) s += kl[c][threadIdx.x];
    Ks[((size_t)((b * NH + h) * NCHUNK + n)) * HD + threadIdx.x] = s;
  }
}

// Phase B: exclusive prefix over chunks. grid (H, B), 256 thr
__global__ __launch_bounds__(256)
void lt_kv_scan(const float* __restrict__ KV, const float* __restrict__ Ks,
                float* __restrict__ KVe, float* __restrict__ Kse) {
  int h = blockIdx.x, b = blockIdx.y;
  size_t base = (size_t)(b * NH + h) * NCHUNK;
  float run[16];
#pragma unroll
  for (int i = 0; i < 16; i++) run[i] = 0.f;
  for (int n = 0; n < NCHUNK; n++) {
    size_t off = (base + n) * (size_t)(HD * HD);
#pragma unroll
    for (int i = 0; i < 16; i++) {
      size_t j = off + threadIdx.x + i * 256;
      KVe[j] = run[i];
      run[i] += KV[j];
    }
  }
  if (threadIdx.x < HD) {
    float r = 0.f;
    for (int n = 0; n < NCHUNK; n++) {
      size_t j = (base + n) * HD + threadIdx.x;
      Kse[j] = r;
      r += Ks[j];
    }
  }
}

// ---------------------------------------------------------------------------
// Attention phase C: out = (q@KVe + tril(q@k^T)@v) / (q@Kse + rowsum + eps)
// grid (nC, H, B), 256 thr (8 waves); WMMA on LDS-staged bf16 tiles.
// ---------------------------------------------------------------------------
__global__ __launch_bounds__(256)
void lt_attn_out(const __bf16* __restrict__ qf, const __bf16* __restrict__ kf,
                 const __bf16* __restrict__ vf, const float* __restrict__ KVe,
                 const float* __restrict__ Kse, __bf16* __restrict__ attb) {
  int n = blockIdx.x, h = blockIdx.y, b = blockIdx.z;
  int s0 = b * SEQ + n * CHUNK;

  __shared__ __align__(16) __bf16 ql[CHUNK][HD];     // q row-major (A)
  __shared__ __align__(16) __bf16 kl[CHUNK][HD];     // k row-major (B for scores)
  __shared__ __align__(16) __bf16 vt[HD][CHUNK];     // v transposed (B for intra)
  __shared__ __align__(16) __bf16 kvt[HD][HD];       // KVe transposed (B for inter)
  __shared__ __align__(16) __bf16 sl[CHUNK][CHUNK];  // masked scores (A for intra)
  __shared__ float den[CHUNK];
  __shared__ float deni[CHUNK];
  __shared__ float ksel[HD];

  size_t kvbase = ((size_t)((b * NH + h) * NCHUNK + n)) * HD * HD;
  size_t ksbase = ((size_t)((b * NH + h) * NCHUNK + n)) * HD;

  for (int e = threadIdx.x; e < CHUNK * HD; e += 256) {
    int r = e >> 6, c = e & 63;
    size_t gi = (size_t)(s0 + r) * D_MODEL + h * HD + c;
    ql[r][c] = qf[gi];
    kl[r][c] = kf[gi];
    vt[c][r] = vf[gi];
  }
  for (int e = threadIdx.x; e < HD * HD; e += 256) {
    int d = e >> 6, ee = e & 63;
    kvt[ee][d] = f2bf(KVe[kvbase + (size_t)d * HD + ee]);
  }
  if (threadIdx.x < HD) ksel[threadIdx.x] = Kse[ksbase + threadIdx.x];
  __syncthreads();

  if (threadIdx.x < CHUNK) {                  // den_inter = q . Kse (per row)
    float s = 0.f;
    for (int d = 0; d < HD; d++) s += bf2f(ql[threadIdx.x][d]) * ksel[d];
    deni[threadIdx.x] = s;
  }

  const int lane = threadIdx.x & 31;
  const int wave = threadIdx.x >> 5;
  const int lr = lane & 15, hf = lane >> 4;
  const int mrow = wave * 16;                 // wave owns 16 rows

  // ---- scores = q @ k^T, masked, bf16 into LDS; den_intra row-sums ----
  float rowsum[8];
#pragma unroll
  for (int i = 0; i < 8; i++) rowsum[i] = 0.f;

  for (int nt = 0; nt < 8; nt++) {
    v8f acc = zero8();
#pragma unroll
    for (int d0 = 0; d0 < HD; d0 += 32) {
      FragU a, bb;
      const __bf16* pa = &ql[mrow + lr][d0 + hf * 8];
      a.u[0] = *(const v4u*)pa;
      a.u[1] = *(const v4u*)(pa + 16);
      const __bf16* pb = &kl[nt * 16 + lr][d0 + hf * 8];
      bb.u[0] = *(const v4u*)pb;
      bb.u[1] = *(const v4u*)(pb + 16);
      acc = __builtin_amdgcn_wmma_f32_16x16x32_bf16(false, a.v, false, bb.v,
                                                    (short)0, acc, false, false);
    }
#pragma unroll
    for (int i = 0; i < 8; i++) {
      int r = mrow + i + hf * 8;              // C layout: row = vgpr + 8*half
      int c = nt * 16 + lr;                   // col = lane & 15 within tile
      float vv = (c <= r) ? acc[i] : 0.f;     // causal (tril, incl. diagonal)
      rowsum[i] += vv;
      sl[r][c] = f2bf(vv);
    }
  }
#pragma unroll
  for (int i = 0; i < 8; i++) {               // reduce across 16-lane row group
    float s = rowsum[i];
    s += __shfl_xor(s, 1, 32);
    s += __shfl_xor(s, 2, 32);
    s += __shfl_xor(s, 4, 32);
    s += __shfl_xor(s, 8, 32);
    if (lr == 0) den[mrow + i + hf * 8] = s;
  }
  __syncthreads();

  // ---- out = scores @ v (K=128) + q @ KVe (K=64), then divide by denom ----
  v8f acc2[4];
#pragma unroll
  for (int nt = 0; nt < 4; nt++) acc2[nt] = zero8();

  for (int nt = 0; nt < 4; nt++) {
#pragma unroll
    for (int c0 = 0; c0 < CHUNK; c0 += 32) {
      FragU a, bb;
      const __bf16* pa = &sl[mrow + lr][c0 + hf * 8];
      a.u[0] = *(const v4u*)pa;
      a.u[1] = *(const v4u*)(pa + 16);
      const __bf16* pb = &vt[nt * 16 + lr][c0 + hf * 8];
      bb.u[0] = *(const v4u*)pb;
      bb.u[1] = *(const v4u*)(pb + 16);
      acc2[nt] = __builtin_amdgcn_wmma_f32_16x16x32_bf16(false, a.v, false, bb.v,
                                                         (short)0, acc2[nt], false, false);
    }
#pragma unroll
    for (int d0 = 0; d0 < HD; d0 += 32) {
      FragU a, bb;
      const __bf16* pa = &ql[mrow + lr][d0 + hf * 8];
      a.u[0] = *(const v4u*)pa;
      a.u[1] = *(const v4u*)(pa + 16);
      const __bf16* pb = &kvt[nt * 16 + lr][d0 + hf * 8];
      bb.u[0] = *(const v4u*)pb;
      bb.u[1] = *(const v4u*)(pb + 16);
      acc2[nt] = __builtin_amdgcn_wmma_f32_16x16x32_bf16(false, a.v, false, bb.v,
                                                         (short)0, acc2[nt], false, false);
    }
  }
#pragma unroll
  for (int nt = 0; nt < 4; nt++)
#pragma unroll
    for (int i = 0; i < 8; i++) {
      int r = mrow + i + hf * 8;
      int e = nt * 16 + lr;
      float o = acc2[nt][i] / (den[r] + deni[r] + 1e-6f);
      attb[(size_t)(s0 + r) * D_MODEL + h * HD + e] = f2bf(o);
    }
}

// ---------------------------------------------------------------------------
// Host-side orchestration
// ---------------------------------------------------------------------------
extern "C" void kernel_launch(void* const* d_in, const int* in_sizes, int n_in,
                              void* d_out, int out_size, void* d_ws, size_t ws_size,
                              hipStream_t stream) {
  const float* x_in = (const float*)d_in[0];
  const float* Wq = (const float*)d_in[1];  const float* bq = (const float*)d_in[2];
  const float* Wk = (const float*)d_in[3];  const float* bk = (const float*)d_in[4];
  const float* Wv = (const float*)d_in[5];  const float* bv = (const float*)d_in[6];
  const float* Wo = (const float*)d_in[7];  const float* bo = (const float*)d_in[8];
  const float* W1 = (const float*)d_in[9];  const float* b1 = (const float*)d_in[10];
  const float* W2 = (const float*)d_in[11]; const float* b2 = (const float*)d_in[12];
  const float* g1 = (const float*)d_in[13]; const float* be1 = (const float*)d_in[14];
  const float* g2 = (const float*)d_in[15]; const float* be2 = (const float*)d_in[16];
  const float* gf = (const float*)d_in[17]; const float* bef = (const float*)d_in[18];

  char* w = (char*)d_ws;
  size_t off = 0;
  auto carve = [&](size_t bytes) -> void* {
    void* p = w + off;
    off = (off + bytes + 255) & ~(size_t)255;
    return p;
  };
  float*  x_f  = (float*) carve((size_t)M_TOT * D_MODEL * 4);
  __bf16* xb   = (__bf16*)carve((size_t)M_TOT * D_MODEL * 2);
  __bf16* qfb  = (__bf16*)carve((size_t)M_TOT * D_MODEL * 2);
  __bf16* kfb  = (__bf16*)carve((size_t)M_TOT * D_MODEL * 2);
  __bf16* vfb  = (__bf16*)carve((size_t)M_TOT * D_MODEL * 2);
  __bf16* attb = (__bf16*)carve((size_t)M_TOT * D_MODEL * 2);
  __bf16* yb   = (__bf16*)carve((size_t)M_TOT * D_FF * 2);
  float*  KV   = (float*) carve((size_t)BATCH * NH * NCHUNK * HD * HD * 4);
  float*  KVe  = (float*) carve((size_t)BATCH * NH * NCHUNK * HD * HD * 4);
  float*  Ks   = (float*) carve((size_t)BATCH * NH * NCHUNK * HD * 4);
  float*  Kse  = (float*) carve((size_t)BATCH * NH * NCHUNK * HD * 4);
  __bf16* WqT  = (__bf16*)carve((size_t)D_MODEL * D_MODEL * 2);
  __bf16* WkT  = (__bf16*)carve((size_t)D_MODEL * D_MODEL * 2);
  __bf16* WvT  = (__bf16*)carve((size_t)D_MODEL * D_MODEL * 2);
  __bf16* WoT  = (__bf16*)carve((size_t)D_MODEL * D_MODEL * 2);
  __bf16* W1T  = (__bf16*)carve((size_t)D_MODEL * D_FF * 2);
  __bf16* W2T  = (__bf16*)carve((size_t)D_FF * D_MODEL * 2);

  const dim3 blk(256);
  size_t nx = (size_t)M_TOT * D_MODEL;
  lt_copy_x<<<dim3((unsigned)((nx + 255) / 256)), blk, 0, stream>>>(x_in, x_f, xb, nx);

  const dim3 gD(D_MODEL / 128, M_TOT / 128);   // (8, 64)
  const dim3 gF(D_FF / 128, M_TOT / 128);      // (32, 64)
  const dim3 gA(NCHUNK, NH, BATCH);            // attention blocks
  const dim3 gS(NH, BATCH);                    // scan blocks
  const dim3 gTsq(D_MODEL / 32, D_MODEL / 32); // transpose grids
  const dim3 gT1(D_FF / 32, D_MODEL / 32);     // W1: K=1024 -> N=4096
  const dim3 gT2(D_MODEL / 32, D_FF / 32);     // W2: K=4096 -> N=1024

  for (int l = 0; l < 4; l++) {
    size_t wOff  = (size_t)l * D_MODEL * D_MODEL;
    size_t w1Off = (size_t)l * D_MODEL * D_FF;
    size_t w2Off = (size_t)l * D_FF * D_MODEL;
    size_t bOff  = (size_t)l * D_MODEL;
    size_t b1Off = (size_t)l * D_FF;

    lt_wt_convert<<<gTsq, blk, 0, stream>>>(Wq + wOff,  WqT, D_MODEL, D_MODEL);
    lt_wt_convert<<<gTsq, blk, 0, stream>>>(Wk + wOff,  WkT, D_MODEL, D_MODEL);
    lt_wt_convert<<<gTsq, blk, 0, stream>>>(Wv + wOff,  WvT, D_MODEL, D_MODEL);
    lt_wt_convert<<<gTsq, blk, 0, stream>>>(Wo + wOff,  WoT, D_MODEL, D_MODEL);
    lt_wt_convert<<<gT1,  blk, 0, stream>>>(W1 + w1Off, W1T, D_MODEL, D_FF);
    lt_wt_convert<<<gT2,  blk, 0, stream>>>(W2 + w2Off, W2T, D_FF, D_MODEL);

    // q,k feature-mapped (elu+1) bf16; v plain bf16
    lt_gemm_bf16<1><<<gD, blk, 0, stream>>>(xb, WqT, bq + bOff, nullptr, nullptr, qfb,
                                            M_TOT, D_MODEL, D_MODEL);
    lt_gemm_bf16<1><<<gD, blk, 0, stream>>>(xb, WkT, bk + bOff, nullptr, nullptr, kfb,
                                            M_TOT, D_MODEL, D_MODEL);
    lt_gemm_bf16<2><<<gD, blk, 0, stream>>>(xb, WvT, bv + bOff, nullptr, nullptr, vfb,
                                            M_TOT, D_MODEL, D_MODEL);

    lt_kv_state<<<gA, blk, 0, stream>>>(kfb, vfb, KV, Ks);
    lt_kv_scan<<<gS, blk, 0, stream>>>(KV, Ks, KVe, Kse);
    lt_attn_out<<<gA, blk, 0, stream>>>(qfb, kfb, vfb, KVe, Kse, attb);

    // x = x + att @ Wo + bo  (residual epilogue)
    lt_gemm_bf16<4><<<gD, blk, 0, stream>>>(attb, WoT, bo + bOff, x_f, x_f, nullptr,
                                            M_TOT, D_MODEL, D_MODEL);
    // x = ln1(x); xb = bf16(x)
    lt_layernorm<<<dim3(M_TOT), blk, 0, stream>>>(x_f, g1 + bOff, be1 + bOff, x_f, xb);
    // y = gelu(x @ W1 + b1) -> bf16
    lt_gemm_bf16<3><<<gF, blk, 0, stream>>>(xb, W1T, b1 + b1Off, nullptr, nullptr, yb,
                                            M_TOT, D_FF, D_MODEL);
    // x = x + y @ W2 + b2
    lt_gemm_bf16<4><<<gD, blk, 0, stream>>>(yb, W2T, b2 + bOff, x_f, x_f, nullptr,
                                            M_TOT, D_MODEL, D_FF);
    // x = ln2(x); xb = bf16(x)
    lt_layernorm<<<dim3(M_TOT), blk, 0, stream>>>(x_f, g2 + bOff, be2 + bOff, x_f, xb);
  }
  // final LayerNorm -> d_out (fp32)
  lt_layernorm<<<dim3(M_TOT), blk, 0, stream>>>(x_f, gf, bef, (float*)d_out, nullptr);

  (void)in_sizes; (void)n_in; (void)out_size; (void)ws_size;
}
